// GATLayer_76622216560853
// MI455X (gfx1250) — matrix operand order repaired and checked by reading, not compiled, not run on hardware
//
#include <hip/hip_runtime.h>
#include <hip/hip_bf16.h>

// ---------------------------------------------------------------------------
// GAT layer, fused for MI455X (gfx1250, wave32, WMMA).
//   B=8, N=2048, F_in=256, D=128
// Stage 0: pack W (fp32 [256,128]) -> bf16 in WMMA B-operand lane layout.
// Stage 1: H_ = H@W via v_wmma_f32_16x16x32_bf16; emit f1,f2 (fp32) and H_
//          as bf16 in the WMMA B-operand layout (L2-resident, 4 MB).
// Stage 2: stream A exactly once (b128 NT loads, software-pipelined one tile
//          ahead), p = exp(masked leaky logit) in fp32, accumulate numerator
//          with bf16 WMMA (2 WMMAs / barrier, double-buffered LDS), divide by
//          the denominator.
// ---------------------------------------------------------------------------

typedef __attribute__((ext_vector_type(16))) __bf16 v16bf;
typedef __attribute__((ext_vector_type(4)))  __bf16 v4bf;
typedef __attribute__((ext_vector_type(2)))  __bf16 v2bf;
typedef __attribute__((ext_vector_type(8)))  float  v8f;
typedef __attribute__((ext_vector_type(4)))  float  v4f;
typedef __attribute__((ext_vector_type(2)))  float  v2f;

#define BATCH 8
#define NNODE 2048
#define FIN   256
#define DDIM  128
#define LEAKY 0.2f

static __device__ __forceinline__ __bf16 f2bf(float x) { return (__bf16)x; }

// ---------------------------------------------------------------------------
// Stage 0: Wb[((kt*8+dt)*32 + lane)*16 + e] = bf16(W[kt*32 + (lane>>4)*16 + e]
//                                                    [dt*16 + (lane&15)])
// so each WMMA lane's 16 B-operand values are 32 contiguous bytes.
// ---------------------------------------------------------------------------
__global__ void pack_w_kernel(const float* __restrict__ W, __bf16* __restrict__ Wb) {
    int idx = blockIdx.x * blockDim.x + threadIdx.x;
    if (idx >= (FIN / 32) * (DDIM / 16) * 32 * 16) return;
    int e  = idx & 15;
    int l  = (idx >> 4) & 31;
    int dt = (idx >> 9) & 7;
    int kt = idx >> 12;
    int krow = kt * 32 + ((l >> 4) << 4) + e;
    int col  = dt * 16 + (l & 15);
    Wb[idx] = f2bf(W[krow * DDIM + col]);
}

// ---------------------------------------------------------------------------
// Stage 1: per block: 16 rows of H_, 8 waves (one 16-wide D-slice each).
// Double-buffered LDS A tile, one barrier per WMMA, H loads pipelined
// one K-tile ahead.
// ---------------------------------------------------------------------------
__global__ void __launch_bounds__(256) gat_proj_kernel(
    const float* __restrict__ H, const __bf16* __restrict__ Wb,
    const float* __restrict__ a1, const float* __restrict__ a2,
    __bf16* __restrict__ Hb, float* __restrict__ f1, float* __restrict__ f2)
{
    __shared__ __attribute__((aligned(32))) __bf16 ldsA[2][32 * 16];
    __shared__ float f1s[16], f2s[16];

    const int tid  = threadIdx.x;
    const int lane = tid & 31;
    const int wv   = tid >> 5;                 // dt: which 16-wide D slice
    const int b    = blockIdx.x >> 7;
    const int i0   = (blockIdx.x & 127) << 4;  // row-tile base

    if (tid < 16) { f1s[tid] = 0.f; f2s[tid] = 0.f; }

    // each thread owns (row m, 2 consecutive k) of the 16x32 tile
    const int m   = tid >> 4;                  // 0..15
    const int kp  = (tid & 15) << 1;           // 0..30 step 2
    const int lgw = m + ((kp >> 4) << 4);
    const int e0  = kp & 15;

    const float* Hrow = H + (size_t)b * NNODE * FIN + (size_t)(i0 + m) * FIN;

    v2f hv = *(const v2f*)&Hrow[kp];           // preload K-tile 0
    v8f acc = {};
    for (int kt = 0; kt < FIN / 32; ++kt) {
        v2f hcur = hv;
        if (kt + 1 < FIN / 32)                 // pipeline next K-tile
            hv = *(const v2f*)&Hrow[(kt + 1) * 32 + kp];
        v2bf pv; pv.x = f2bf(hcur.x); pv.y = f2bf(hcur.y);
        *(v2bf*)&ldsA[kt & 1][lgw * 16 + e0] = pv;   // ds_store_b32
        __syncthreads();
        v16bf amat = *(const v16bf*)&ldsA[kt & 1][lane * 16];
        v16bf bmat = *(const v16bf*)&Wb[((kt * 8 + wv) * 32 + lane) * 16];
        acc = __builtin_amdgcn_wmma_f32_16x16x32_bf16(false, amat, false, bmat,
                                                      (short)0, acc, false, false);
    }

    // acc[i] = H_[i0 + r][d], r = (lane>>4)*8 + i, d = wv*16 + (lane&15)
    const int n     = lane & 15;
    const int rbase = (lane >> 4) << 3;
    const int d     = wv * 16 + n;
    const float a1d = a1[d], a2d = a2[d];

    #pragma unroll
    for (int i = 0; i < 8; ++i) {
        int r = rbase + i;
        atomicAdd(&f1s[r], acc[i] * a1d);
        atomicAdd(&f2s[r], acc[i] * a2d);
        int j  = i0 + r;
        int jt = j >> 5, k = j & 31;
        size_t flat = ((((size_t)b * (NNODE / 32) + jt) * 8 + wv) * 32 +
                       (((k >> 4) << 4) | n)) * 16 + (k & 15);
        Hb[flat] = f2bf(acc[i]);   // swizzled bf16 B-operand layout
    }
    __syncthreads();
    if (tid < 16) {
        f1[(size_t)b * NNODE + i0 + tid] = f1s[tid];
        f2[(size_t)b * NNODE + i0 + tid] = f2s[tid];
    }
}

// ---------------------------------------------------------------------------
// Stage 2: stream A once; fused mask+leaky+exp+softmax+alpha@H_.
// One block = 16 output rows; 8 waves each own one 16-wide D slice.
// 64 columns per iteration: 1 b128 NT load of A, 1 ds_store_b64, 1 barrier,
// 2 WMMAs. A/f2/Hb loads software-pipelined one tile ahead; double-buffered
// LDS P tile.
// ---------------------------------------------------------------------------
__global__ void __launch_bounds__(256) gat_attn_kernel(
    const float* __restrict__ A, const __bf16* __restrict__ Hb,
    const float* __restrict__ f1, const float* __restrict__ f2,
    float* __restrict__ out)
{
    __shared__ __attribute__((aligned(32))) __bf16 ldsP[2][2 * 32 * 16];
    __shared__ float lsum[16];

    const int tid  = threadIdx.x;
    const int lane = tid & 31;
    const int wv   = tid >> 5;                 // dt
    const int b    = blockIdx.x >> 7;
    const int i0   = (blockIdx.x & 127) << 4;

    if (tid < 16) lsum[tid] = 0.f;

    // each thread owns (row m, 4 consecutive k) of the 16x64 P tile
    const int m    = tid >> 4;                 // 0..15
    const int k4   = (tid & 15) << 2;          // 0..60 step 4
    const int sub  = k4 >> 5;                  // which 16x32 half
    const int kl   = k4 & 31;
    const int lgw  = m + ((kl >> 4) << 4);
    const int e0   = kl & 15;
    const int pidx = sub * 512 + lgw * 16 + e0;

    const float   f1v  = f1[(size_t)b * NNODE + i0 + m];
    const float*  f2b  = f2 + (size_t)b * NNODE;
    const float*  Arow = A + (size_t)b * NNODE * NNODE + (size_t)(i0 + m) * NNODE;
    const __bf16* Hbb  = Hb + (size_t)b * NNODE * DDIM;
    const __bf16* Hlane = Hbb + (size_t)(wv * 32 + lane) * 16;  // + jt32*4096

    // preload tile 0 (A stream NT so it never evicts the L2-resident Hb)
    v4f   av = __builtin_nontemporal_load((const v4f*)&Arow[k4]);
    v4f   fv = *(const v4f*)&f2b[k4];
    v16bf h0 = *(const v16bf*)&Hlane[0];
    v16bf h1 = *(const v16bf*)&Hlane[4096];

    float den = 0.f;
    v8f acc = {};

    for (int jt = 0; jt < NNODE / 64; ++jt) {
        v4f   avc = av, fvc = fv;
        v16bf h0c = h0, h1c = h1;
        if (jt + 1 < NNODE / 64) {             // pipeline next 64-col tile
            const int jn = (jt + 1) * 64;
            av = __builtin_nontemporal_load((const v4f*)&Arow[jn + k4]);
            fv = *(const v4f*)&f2b[jn + k4];
            h0 = *(const v16bf*)&Hlane[(size_t)(jt + 1) * 8192];
            h1 = *(const v16bf*)&Hlane[(size_t)(jt + 1) * 8192 + 4096];
        }
        if (jt + 3 < NNODE / 64)               // uniform prefetch further out
            __builtin_prefetch(&Arow[(jt + 3) * 64 + k4], 0, 0);

        v4bf pb;
        #pragma unroll
        for (int c = 0; c < 4; ++c) {
            float s = f1v + fvc[c];
            s = (s >= 0.f) ? s : LEAKY * s;
            // masked logit is 0 -> exp=1 (reference multiplies logits by mask)
            float p = (avc[c] > 0.f) ? __expf(s) : 1.0f;
            den  += p;
            pb[c] = f2bf(p);
        }
        *(v4bf*)&ldsP[jt & 1][pidx] = pb;      // ds_store_b64
        __syncthreads();
        const __bf16* pbase = &ldsP[jt & 1][lane * 16];
        v16bf p0 = *(const v16bf*)pbase;
        v16bf p1 = *(const v16bf*)(pbase + 512);
        acc = __builtin_amdgcn_wmma_f32_16x16x32_bf16(false, p0, false, h0c,
                                                      (short)0, acc, false, false);
        acc = __builtin_amdgcn_wmma_f32_16x16x32_bf16(false, p1, false, h1c,
                                                      (short)0, acc, false, false);
    }

    atomicAdd(&lsum[m], den);                  // 16 partials per row
    __syncthreads();

    const int n     = lane & 15;
    const int rbase = (lane >> 4) << 3;
    float* ob = out + ((size_t)b * NNODE + i0) * DDIM + wv * 16 + n;
    #pragma unroll
    for (int i = 0; i < 8; ++i) {
        int r = rbase + i;
        ob[(size_t)r * DDIM] = acc[i] / lsum[r];
    }
}

// ---------------------------------------------------------------------------
extern "C" void kernel_launch(void* const* d_in, const int* in_sizes, int n_in,
                              void* d_out, int out_size, void* d_ws, size_t ws_size,
                              hipStream_t stream) {
    const float* A  = (const float*)d_in[0];   // [8,2048,2048]
    const float* H  = (const float*)d_in[1];   // [8,2048,256]
    const float* W  = (const float*)d_in[2];   // [256,128]
    const float* a1 = (const float*)d_in[3];   // [128,1]
    const float* a2 = (const float*)d_in[4];   // [128,1]
    float* out = (float*)d_out;                // [8,2048,128] fp32

    // workspace: Wb 64KB | f1 64KB | f2 64KB | Hb 4MB
    char*   ws = (char*)d_ws;
    __bf16* Wb = (__bf16*)ws;
    float*  f1 = (float*)(ws + (64 << 10));
    float*  f2 = (float*)(ws + (128 << 10));
    __bf16* Hb = (__bf16*)(ws + (192 << 10));

    (void)in_sizes; (void)n_in; (void)out_size; (void)ws_size;

    const int packN = (FIN / 32) * (DDIM / 16) * 32 * 16;       // 32768
    pack_w_kernel<<<packN / 256, 256, 0, stream>>>(W, Wb);
    gat_proj_kernel<<<BATCH * (NNODE / 16), 256, 0, stream>>>(H, Wb, a1, a2, Hb, f1, f2);
    gat_attn_kernel<<<BATCH * (NNODE / 16), 256, 0, stream>>>(A, Hb, f1, f2, out);
}
